// EncoderLayer_76785425317989
// MI455X (gfx1250) — compile-verified, matrix operand order checked by reading, not run on hardware
//
#include <hip/hip_runtime.h>
#include <hip/hip_bf16.h>

// ---------------- Problem constants ----------------
#define BB   8
#define LL   2048
#define DD   512
#define KAV  25
#define PAV  12
#define EPSV 1e-5f
#define SCALE_QK 0.044194173824159216f   // 1/sqrt(512)

typedef __bf16 bf16;
typedef bf16  bf16x16 __attribute__((ext_vector_type(16)));
typedef float f32x8   __attribute__((ext_vector_type(8)));
typedef unsigned int uint32x4 __attribute__((ext_vector_type(4)));
typedef unsigned int uint32x8 __attribute__((ext_vector_type(8)));

__device__ __forceinline__ bf16  f2bf(float f) { return (bf16)f; }
__device__ __forceinline__ float bf2f(bf16 h)  { return (float)h; }

__device__ __forceinline__ f32x8 wmma_bf16(bf16x16 a, bf16x16 b, f32x8 c) {
    // D = A(16x32 bf16) * B(32x16 bf16) + C(16x16 f32)
    return __builtin_amdgcn_wmma_f32_16x16x32_bf16(
        /*neg_a=*/false, a, /*neg_b=*/false, b,
        /*c_mod=*/(short)0, c, /*reuse_a=*/false, /*reuse_b=*/false);
}

// Load one 16x32 bf16 fragment ("row-major per lane" layout, ISA 7.12.2):
// lane L in 0..15 holds row M=L with K = {0..7, 16..23}; lane L+16 holds
// the same row with K = {8..15, 24..31}. Two 16B LDS reads per lane.
__device__ __forceinline__ bf16x16 ld_frag(const bf16* p, int stride) {
    int lane = threadIdx.x & 31;
    int lm = lane & 15, hi = lane >> 4;
    const bf16* r = p + lm * stride + 8 * hi;
    union { bf16x16 v; uint4 u[2]; } f;
    f.u[0] = *(const uint4*)(r);
    f.u[1] = *(const uint4*)(r + 16);
    return f.v;
}

// ---------------- Tensor Data Mover: 2D bf16 tile, Global -> LDS ----------------
// Builds a D# per CDNA5 ISA ch.8 (group0 128b + group1 256b) and issues
// TENSOR_LOAD_TO_LDS. Tracked by TENSORcnt; caller must s_wait_tensorcnt
// and barrier before consuming the LDS tile. Issue from ONE wave per block.
//   tile: tile_d1 rows x tile_d0 bf16 elements, row stride = stride0 elements
__device__ __forceinline__ void tdm_load_2d(const void* lds_ptr, const void* gptr,
                                            unsigned tensor_d0, unsigned tensor_d1,
                                            unsigned tile_d0, unsigned tile_d1,
                                            unsigned stride0) {
    unsigned long long ga = (unsigned long long)(uintptr_t)gptr;
    unsigned lds  = (unsigned)__builtin_amdgcn_readfirstlane((int)(uintptr_t)lds_ptr);
    unsigned galo = (unsigned)__builtin_amdgcn_readfirstlane((int)(ga & 0xFFFFFFFFull));
    unsigned gahi = (unsigned)__builtin_amdgcn_readfirstlane((int)(ga >> 32));

    uint32x4 g0;
    g0[0] = 1u;                                   // count=1 (valid), no gather
    g0[1] = lds;                                  // lds_addr [63:32]
    g0[2] = galo;                                 // global_addr [95:64]
    g0[3] = (gahi & 0x01FFFFFFu) | 0x80000000u;   // global_addr[56:32] | type=2

    uint32x8 g1;
    g1[0] = 1u << 16;                             // workgroup_mask=0, data_size=2B
    g1[1] = (tensor_d0 & 0xFFFFu) << 16;          // tensor_dim0 [79:48] low half
    g1[2] = (tensor_d0 >> 16) | ((tensor_d1 & 0xFFFFu) << 16); // dim0 hi | dim1 lo
    g1[3] = (tensor_d1 >> 16) | (tile_d0 << 16);  // dim1 hi | tile_dim0 [127:112]
    g1[4] = tile_d1;                              // tile_dim1, tile_dim2=0
    g1[5] = stride0;                              // tensor_dim0_stride low 32
    g1[6] = 0u;
    g1[7] = 0u;

    asm volatile("tensor_load_to_lds %0, %1" :: "s"(g0), "s"(g1) : "memory");
}

__device__ __forceinline__ void tdm_wait0() {
    __builtin_amdgcn_s_wait_tensorcnt(0);
}

// ---------------- 1. fp32 -> bf16 convert ----------------
__global__ void cvt_kernel(const float* __restrict__ in, bf16* __restrict__ out, int n) {
    int i = blockIdx.x * 256 + threadIdx.x;
    if (i < n) out[i] = f2bf(in[i]);
}

// ---------------- 2. series decomposition (AvgPool K=25, pad=12) ----------------
// One thread per (b, d) column, running window sum over L.
__global__ void decomp_kernel(const float* __restrict__ x,
                              float* __restrict__ trend,
                              float* __restrict__ seasF,
                              bf16*  __restrict__ seasB) {
    int b = blockIdx.y;
    int d = blockIdx.x * 256 + threadIdx.x;           // gridDim.x = DD/256
    const float* xp = x + (size_t)b * LL * DD + d;
    float sum = 0.f;
    for (int l = 0; l <= PAV; ++l) sum += xp[(size_t)l * DD];
    const float invK = 1.0f / (float)KAV;
    for (int l = 0; l < LL; ++l) {
        if ((l & 15) == 0 && l + 64 < LL)
            __builtin_prefetch(xp + (size_t)(l + 64) * DD, 0, 1);
        float t  = sum * invK;
        float xv = xp[(size_t)l * DD];
        float s  = xv - t;
        size_t idx = (size_t)b * LL * DD + (size_t)l * DD + d;
        trend[idx] = t;
        seasF[idx] = s;
        seasB[idx] = f2bf(s);
        int add = l + PAV + 1; if (add < LL) sum += xp[(size_t)add * DD];
        int sub = l - PAV;     if (sub >= 0) sum -= xp[(size_t)sub * DD];
    }
}

// ---------------- 3. WMMA GEMM: out[M,512] = A[M,512](bf16) @ W[512,512](bf16) + bias ----------------
// Block tile 128x128, 256 threads = 8 waves (4x2), wave tile 32x64.
// A tile staged by the Tensor Data Mover; W tile transposed manually.
__global__ __launch_bounds__(256)
void gemm_kernel(const bf16* __restrict__ A, const bf16* __restrict__ W,
                 const float* __restrict__ bias,
                 float* __restrict__ outF, bf16* __restrict__ outB,
                 int act /*0=none,1=relu*/) {
    __shared__ __align__(16) bf16 sA [128 * 32];   // [m][k]
    __shared__ __align__(16) bf16 sBt[128 * 32];   // [n][k]  (transposed W tile)

    int m0 = blockIdx.x * 128, n0 = blockIdx.y * 128;
    int tid = threadIdx.x, wave = tid >> 5, lane = tid & 31;
    int lm = lane & 15, hi = lane >> 4;
    int wm = wave & 3, wn = wave >> 2;

    f32x8 acc[2][4] = {};

    for (int k0 = 0; k0 < 512; k0 += 32) {
        // TDM: A tile (128 rows x 32 bf16, row stride 512) -> sA
        if (wave == 0)
            tdm_load_2d(sA, A + (size_t)m0 * 512 + k0,
                        /*tensor_d0=*/512, /*tensor_d1=*/16384,
                        /*tile_d0=*/32, /*tile_d1=*/128, /*stride0=*/512);
        { // stage W tile transposed: read 16 bf16 along n, scatter to sBt[n][k]
            int kr = tid >> 3, nc = (tid & 7) << 4;
            const uint4* gp = (const uint4*)(W + (size_t)(k0 + kr) * 512 + n0 + nc);
            uint4 p0 = gp[0], p1 = gp[1];
            const bf16* pp = (const bf16*)&p0;
            for (int j = 0; j < 8; ++j) sBt[(nc + j) * 32 + kr] = pp[j];
            pp = (const bf16*)&p1;
            for (int j = 0; j < 8; ++j) sBt[(nc + 8 + j) * 32 + kr] = pp[j];
        }
        if (wave == 0) tdm_wait0();
        __syncthreads();
        for (int mi = 0; mi < 2; ++mi) {
            bf16x16 afr = ld_frag(sA + (wm * 32 + mi * 16) * 32, 32);
            for (int ni = 0; ni < 4; ++ni) {
                bf16x16 bfr = ld_frag(sBt + (wn * 64 + ni * 16) * 32, 32);
                acc[mi][ni] = wmma_bf16(afr, bfr, acc[mi][ni]);
            }
        }
        __syncthreads();
    }

    for (int mi = 0; mi < 2; ++mi)
        for (int ni = 0; ni < 4; ++ni) {
            int col = n0 + wn * 64 + ni * 16 + lm;
            float bv = bias ? bias[col] : 0.f;
            for (int r = 0; r < 8; ++r) {
                int row = m0 + wm * 32 + mi * 16 + r + 8 * hi;
                float v = acc[mi][ni][r] + bv;
                if (act) v = fmaxf(v, 0.f);
                size_t idx = (size_t)row * 512 + col;
                if (outF) outF[idx] = v;
                if (outB) outB[idx] = f2bf(v);
            }
        }
}

// ---------------- 4. Flash attention: O = softmax(Q K^T / sqrt(D)) V ----------------
// grid = (L/64, B), block = 256 (8 waves). 64-query tile per block; stream
// 64-key tiles; ~217KB dynamic LDS (one workgroup per 320KB WGP).
// Q and K tiles staged by the Tensor Data Mover; V transposed manually.
__global__ __launch_bounds__(256)
void fa_kernel(const bf16* __restrict__ Q, const bf16* __restrict__ Km,
               const bf16* __restrict__ V, bf16* __restrict__ O) {
    extern __shared__ char smem[];
    bf16*  sQ  = (bf16*)smem;             // [64][512]
    bf16*  sK  = sQ  + 64 * 512;          // [64][512]
    bf16*  sVt = sK  + 64 * 512;          // [512][64]  (V transposed: dim-major)
    float* sS  = (float*)(sVt + 512 * 64);// [64][64]
    bf16*  sP  = (bf16*)(sS + 64 * 64);   // [64][64]
    float* sM  = (float*)(sP + 64 * 64);  // [64]
    float* sL2 = sM + 64;                 // [64]
    float* sF  = sL2 + 64;                // [64]

    int b = blockIdx.y;
    int q0 = blockIdx.x * 64;
    int tid = threadIdx.x, wave = tid >> 5, lane = tid & 31;
    int lm = lane & 15, hi = lane >> 4;
    const size_t bb = (size_t)b * (LL * DD);

    // TDM: Q tile (64 x 512 bf16, contiguous rows) -> sQ
    if (wave == 0)
        tdm_load_2d(sQ, Q + bb + (size_t)q0 * DD,
                    512, LL, 512, 64, 512);
    if (tid < 64) { sM[tid] = -3.0e38f; sL2[tid] = 0.f; }

    f32x8 o[4][4] = {};   // wave owns cols [wave*64, wave*64+64), all 64 rows
    if (wave == 0) tdm_wait0();
    __syncthreads();

    for (int kt = 0; kt < LL / 64; ++kt) {
        const size_t kb = bb + (size_t)kt * 64 * DD;
        // TDM: K tile -> sK
        if (wave == 0)
            tdm_load_2d(sK, Km + kb, 512, LL, 512, 64, 512);
        // V tile transposed into sVt (dim-major) — needs a real transpose.
        for (int i = tid; i < 64 * 512 / 8; i += 256) {
            int r  = i >> 6;            // key row 0..63
            int c8 = (i & 63) << 3;     // dim start
            uint4 pack = ((const uint4*)(V + kb))[i];
            const bf16* pv = (const bf16*)&pack;
            for (int j = 0; j < 8; ++j) sVt[(size_t)(c8 + j) * 64 + r] = pv[j];
        }
        if (wave == 0) tdm_wait0();
        __syncthreads();

        // S = Q K^T * scale : 16 tiles of 16x16, 2 per wave, K-dim = 512
        for (int t = wave * 2; t < wave * 2 + 2; ++t) {
            int mi = t >> 2, ni = t & 3;
            f32x8 s = {};
            for (int kk = 0; kk < 16; ++kk) {
                bf16x16 afr = ld_frag(sQ + (mi * 16) * 512 + kk * 32, 512);
                bf16x16 bfr = ld_frag(sK + (ni * 16) * 512 + kk * 32, 512);
                s = wmma_bf16(afr, bfr, s);
            }
            for (int r = 0; r < 8; ++r) {
                int row = mi * 16 + r + 8 * hi;
                sS[row * 64 + ni * 16 + lm] = s[r] * SCALE_QK;
            }
        }
        __syncthreads();

        // online softmax (one thread per query row)
        if (tid < 64) {
            int r = tid;
            float mo = sM[r], mx = mo;
            for (int j = 0; j < 64; ++j) mx = fmaxf(mx, sS[r * 64 + j]);
            float sum = 0.f;
            for (int j = 0; j < 64; ++j) {
                float p = __expf(sS[r * 64 + j] - mx);
                sP[r * 64 + j] = f2bf(p);
                sum += p;
            }
            float fac = __expf(mo - mx);
            sL2[r] = sL2[r] * fac + sum;
            sM[r] = mx;
            sF[r] = fac;
        }
        __syncthreads();

        // rescale running output, then O += P @ V  (K-dim = 64 keys)
        for (int mi = 0; mi < 4; ++mi) {
            float f[8];
            for (int r = 0; r < 8; ++r) f[r] = sF[mi * 16 + r + 8 * hi];
            for (int nt = 0; nt < 4; ++nt)
                for (int r = 0; r < 8; ++r) o[mi][nt][r] *= f[r];
        }
        for (int ks = 0; ks < 2; ++ks)
            for (int mi = 0; mi < 4; ++mi) {
                bf16x16 afr = ld_frag(sP + (mi * 16) * 64 + ks * 32, 64);
                for (int nt = 0; nt < 4; ++nt) {
                    bf16x16 bfr = ld_frag(sVt + (wave * 64 + nt * 16) * 64 + ks * 32, 64);
                    o[mi][nt] = wmma_bf16(afr, bfr, o[mi][nt]);
                }
            }
        __syncthreads();
    }

    // normalize and store bf16
    for (int mi = 0; mi < 4; ++mi) {
        float inv[8];
        for (int r = 0; r < 8; ++r) inv[r] = 1.0f / sL2[mi * 16 + r + 8 * hi];
        for (int nt = 0; nt < 4; ++nt) {
            int col = wave * 64 + nt * 16 + lm;
            for (int r = 0; r < 8; ++r) {
                int grow = q0 + mi * 16 + r + 8 * hi;
                O[bb + (size_t)grow * DD + col] = f2bf(o[mi][nt][r] * inv[r]);
            }
        }
    }
}

// ---------------- 5. LayerNorm (+residual, +addend) per token row ----------------
__global__ __launch_bounds__(256)
void ln_kernel(const float* __restrict__ a, const float* __restrict__ b,
               const float* __restrict__ gamma, const float* __restrict__ beta,
               const float* __restrict__ addend,
               float* __restrict__ outF, bf16* __restrict__ outB) {
    int row = blockIdx.x, tid = threadIdx.x;
    size_t base = (size_t)row * DD;
    float v0 = a[base + tid]       + (b ? b[base + tid]       : 0.f);
    float v1 = a[base + tid + 256] + (b ? b[base + tid + 256] : 0.f);

    __shared__ float r1[256], r2[256];
    r1[tid] = v0 + v1;
    r2[tid] = v0 * v0 + v1 * v1;
    __syncthreads();
    for (int s = 128; s > 0; s >>= 1) {
        if (tid < s) { r1[tid] += r1[tid + s]; r2[tid] += r2[tid + s]; }
        __syncthreads();
    }
    float mean = r1[0] * (1.0f / DD);
    float var  = r2[0] * (1.0f / DD) - mean * mean;
    float rstd = rsqrtf(var + EPSV);

    float o0 = (v0 - mean) * rstd * gamma[tid]       + beta[tid];
    float o1 = (v1 - mean) * rstd * gamma[tid + 256] + beta[tid + 256];
    if (addend) { o0 += addend[base + tid]; o1 += addend[base + tid + 256]; }
    if (outF) { outF[base + tid] = o0; outF[base + tid + 256] = o1; }
    if (outB) { outB[base + tid] = f2bf(o0); outB[base + tid + 256] = f2bf(o1); }
}

// ---------------- host orchestration ----------------
extern "C" void kernel_launch(void* const* d_in, const int* in_sizes, int n_in,
                              void* d_out, int out_size, void* d_ws, size_t ws_size,
                              hipStream_t stream) {
    const float* x   = (const float*)d_in[0];
    const float* w_f[6]  = { (const float*)d_in[1], (const float*)d_in[2],
                             (const float*)d_in[3], (const float*)d_in[4],
                             (const float*)d_in[5], (const float*)d_in[6] };
    const float* bq  = (const float*)d_in[7];
    const float* bk  = (const float*)d_in[8];
    const float* bv  = (const float*)d_in[9];
    const float* bo  = (const float*)d_in[10];
    const float* bb1 = (const float*)d_in[11];
    const float* bb2 = (const float*)d_in[12];
    const float* g1  = (const float*)d_in[13];
    const float* be1 = (const float*)d_in[14];
    const float* g2  = (const float*)d_in[15];
    const float* be2 = (const float*)d_in[16];

    const size_t NBL = (size_t)BB * LL * DD;   // 8,388,608 elements

    char* w = (char*)d_ws;
    auto alloc = [&](size_t bytes) { char* p = w; w += (bytes + 255) & ~(size_t)255; return p; };
    float* trend = (float*)alloc(NBL * 4);
    float* seasF = (float*)alloc(NBL * 4);
    bf16*  seasB = (bf16*) alloc(NBL * 2);
    bf16*  Qb    = (bf16*) alloc(NBL * 2);
    bf16*  Kb    = (bf16*) alloc(NBL * 2);
    bf16*  Vb    = (bf16*) alloc(NBL * 2);
    bf16*  AOb   = (bf16*) alloc(NBL * 2);     // attention output (bf16)
    float* xat   = (float*)alloc(NBL * 4);     // x_attn, reused for x_ff
    float* h1F   = (float*)alloc(NBL * 4);
    bf16*  h1B   = (bf16*) alloc(NBL * 2);
    bf16*  tB    = (bf16*) alloc(NBL * 2);     // relu(h1@w1+bb1) in bf16
    bf16*  wB[6];
    for (int i = 0; i < 6; ++i) wB[i] = (bf16*)alloc((size_t)DD * DD * 2);

    // 1. convert weights to bf16
    for (int i = 0; i < 6; ++i)
        cvt_kernel<<<(DD * DD + 255) / 256, 256, 0, stream>>>(w_f[i], wB[i], DD * DD);

    // 2. series decomposition
    decomp_kernel<<<dim3(DD / 256, BB), 256, 0, stream>>>(x, trend, seasF, seasB);

    // 3. Q/K/V projections
    dim3 gg(16384 / 128, 512 / 128);
    gemm_kernel<<<gg, 256, 0, stream>>>(seasB, wB[0], bq, nullptr, Qb, 0);
    gemm_kernel<<<gg, 256, 0, stream>>>(seasB, wB[1], bk, nullptr, Kb, 0);
    gemm_kernel<<<gg, 256, 0, stream>>>(seasB, wB[2], bv, nullptr, Vb, 0);

    // 4. flash attention
    const size_t fa_lds = (size_t)(3 * 64 * 512 * 2) + 64 * 64 * 4 + 64 * 64 * 2 + 3 * 64 * 4;
    (void)hipFuncSetAttribute((const void*)fa_kernel,
                              hipFuncAttributeMaxDynamicSharedMemorySize, (int)fa_lds);
    fa_kernel<<<dim3(LL / 64, BB), 256, fa_lds, stream>>>(Qb, Kb, Vb, AOb);

    // 5. output projection + LN1
    gemm_kernel<<<gg, 256, 0, stream>>>(AOb, wB[3], bo, xat, nullptr, 0);
    ln_kernel<<<BB * LL, 256, 0, stream>>>(seasF, xat, g1, be1, nullptr, h1F, h1B);

    // 6. feed-forward + LN2 (+trend)
    gemm_kernel<<<gg, 256, 0, stream>>>(h1B, wB[4], bb1, nullptr, tB, 1);
    gemm_kernel<<<gg, 256, 0, stream>>>(tB, wB[5], bb2, xat, nullptr, 0);
    ln_kernel<<<BB * LL, 256, 0, stream>>>(h1F, xat, g2, be2, trend, (float*)d_out, nullptr);
}
